// PGCN_62156766707826
// MI455X (gfx1250) — matrix operand order repaired and checked by reading, not compiled, not run on hardware
//
#include <hip/hip_runtime.h>
#include <hip/hip_bf16.h>

typedef __bf16 bf16_t;
typedef __attribute__((ext_vector_type(16))) __bf16 v16bf;
typedef __attribute__((ext_vector_type(8)))  float  v8f;

// ---------------- elementwise / graph-prep kernels ----------------

__global__ void k_fill(float* p, float v, int n) {
    int i = blockIdx.x * blockDim.x + threadIdx.x;
    if (i < n) p[i] = v;
}

__global__ void k_count_deg(const int* __restrict__ dst, float* __restrict__ deg, int E) {
    int e = blockIdx.x * blockDim.x + threadIdx.x;
    if (e < E) atomicAdd(&deg[dst[e]], 1.0f);
}

// deg (in-place -> self_norm = 2*dinv^2), dinv written separately
__global__ void k_norm(float* __restrict__ deg_selfnorm, float* __restrict__ dinv, int n) {
    int i = blockIdx.x * blockDim.x + threadIdx.x;
    if (i < n) {
        float di = rsqrtf(deg_selfnorm[i]);   // deg >= 2, always positive
        dinv[i] = di;
        deg_selfnorm[i] = 2.0f * di * di;
    }
}

__global__ void k_edge_norm(const int* __restrict__ ei, const float* __restrict__ dinv,
                            float* __restrict__ norm_e, int E) {
    int e = blockIdx.x * blockDim.x + threadIdx.x;
    if (e < E) norm_e[e] = dinv[ei[e]] * dinv[ei[E + e]];
}

__global__ void k_tanh(const float* __restrict__ in, float* __restrict__ out, int n) {
    int i = blockIdx.x * blockDim.x + threadIdx.x;
    if (i < n) out[i] = tanhf(in[i]);
}

// agg[n,f] = self_norm[n]*t[n,f] + b[f]
template<int NC>
__global__ void k_agg_init(const float* __restrict__ t, const float* __restrict__ self_norm,
                           const float* __restrict__ b, float* __restrict__ agg, int total) {
    int i = blockIdx.x * blockDim.x + threadIdx.x;
    if (i < total) {
        int n = i / NC, f = i % NC;
        agg[i] = self_norm[n] * t[i] + b[f];
    }
}

// one edge per (NC/4) lanes; float4 gather from t[src], 4 f32 atomics into agg[dst]
template<int NC>
__global__ void k_scatter(const float* __restrict__ t, const int* __restrict__ ei,
                          const float* __restrict__ norm_e, float* __restrict__ agg, int E) {
    constexpr int TPE = NC / 4;        // threads per edge
    constexpr int EPB = 256 / TPE;     // edges per 256-thread block
    int e = blockIdx.x * EPB + (int)(threadIdx.x / TPE);
    int f = (threadIdx.x % TPE) * 4;
    if (e >= E) return;
    int s = ei[e];
    int d = ei[E + e];
    float w = norm_e[e];
    const float4 v = *reinterpret_cast<const float4*>(t + (long)s * NC + f);
    float* o = agg + (long)d * NC + f;
    atomicAdd(o + 0, w * v.x);
    atomicAdd(o + 1, w * v.y);
    atomicAdd(o + 2, w * v.z);
    atomicAdd(o + 3, w * v.w);
}

// ---------------- fragment packing ----------------
// CDNA5 wave32 WMMA bf16 fragment layouts (ISA 7.12.2):
//   A (16x32, M x K):  lane half h, element e, v=e>>1 -> K = 2*((v&3) + 4h + 8*(v>>2)) + (e&1)
//   B (32x16, K x N):  lane half h, element e          -> K = e + 8h + 8*(e>>3)
// Packed order: frag[((tile * (K/32) + kt) * 32 + lane) * 16 + e], 32 B per lane, coalesced.

__device__ __forceinline__ int a_kidx(int e, int half) {
    int v = e >> 1;
    return 2 * ((v & 3) + 4 * half + 8 * (v >> 2)) + (e & 1);
}

// pack activations [M,256] f32 (optionally ReLU'd) into A-fragment order, bf16
__global__ void k_pack_a(const float* __restrict__ h, bf16_t* __restrict__ Apack,
                         int total_threads /* = (M/16)*8*32 */, int do_relu) {
    int tid = blockIdx.x * blockDim.x + threadIdx.x;
    if (tid >= total_threads) return;
    int lane = tid & 31;
    int kt   = (tid >> 5) & 7;
    int rt   = tid >> 8;
    int half = lane >> 4;
    int l16  = lane & 15;
    const float* src = h + ((long)rt * 16 + l16) * 256 + kt * 32;
    v16bf a;
    #pragma unroll
    for (int e = 0; e < 16; ++e) {
        float f = src[a_kidx(e, half)];
        if (do_relu) f = fmaxf(f, 0.0f);
        a[e] = (bf16_t)f;
    }
    *reinterpret_cast<v16bf*>(Apack + (long)tid * 16) = a;
}

// pack weights [256,NC] f32 into B-fragment order, bf16
__global__ void k_pack_b(const float* __restrict__ W, bf16_t* __restrict__ Bpack,
                         int NC, int total_threads /* = (NC/16)*8*32 */) {
    int tid = blockIdx.x * blockDim.x + threadIdx.x;
    if (tid >= total_threads) return;
    int lane = tid & 31;
    int kt   = (tid >> 5) & 7;
    int ct   = tid >> 8;
    int half = lane >> 4;
    int l16  = lane & 15;
    v16bf b;
    #pragma unroll
    for (int e = 0; e < 16; ++e) {
        int k = kt * 32 + e + 8 * half + 8 * (e >> 3);
        b[e] = (bf16_t)W[(long)k * NC + ct * 16 + l16];
    }
    *reinterpret_cast<v16bf*>(Bpack + (long)tid * 16) = b;
}

// ---------------- WMMA GEMM on packed fragments (no LDS, no barriers) ----------------
// C[M x NC] = A[M x 256] * B[256 x NC]; each wave owns one 16x16 tile,
// streaming coalesced 32B fragments from L2 and issuing 8 v_wmma per tile.
template<int NC, int WAVES>
__global__ __launch_bounds__(32 * WAVES)
void gemm_wmma(const bf16_t* __restrict__ Apack, const bf16_t* __restrict__ Bpack,
               float* __restrict__ C)
{
    constexpr int KT = 256 / 32;               // 8 K-steps
    const int tid  = threadIdx.x;
    const int wave = tid >> 5;
    const int lane = tid & 31;
    const int half = lane >> 4;
    const int l16  = lane & 15;
    const int rt   = blockIdx.x;               // row tile
    const int ct   = blockIdx.y * WAVES + wave;// col tile

    const v16bf* __restrict__ Ap =
        reinterpret_cast<const v16bf*>(Apack + (long)rt * KT * 32 * 16);
    const v16bf* __restrict__ Bp =
        reinterpret_cast<const v16bf*>(Bpack + (long)ct * KT * 32 * 16);

    v8f acc = {};
    #pragma unroll
    for (int kt = 0; kt < KT; ++kt) {
        v16bf a = Ap[kt * 32 + lane];
        v16bf b = Bp[kt * 32 + lane];
        acc = __builtin_amdgcn_wmma_f32_16x16x32_bf16(
            /*neg_a=*/false, a, /*neg_b=*/false, b,
            /*c_mod=*/(short)0, acc, /*reuse_a=*/false, /*reuse_b=*/false);
    }

    // C/D layout: VGPR r -> M = r + 8*half, N = l16
    #pragma unroll
    for (int r = 0; r < 8; ++r) {
        long row = (long)rt * 16 + r + 8 * half;
        C[row * NC + ct * 16 + l16] = acc[r];
    }
}

// ---------------- host-side orchestration ----------------

static inline int cdiv(long a, long b) { return (int)((a + b - 1) / b); }

extern "C" void kernel_launch(void* const* d_in, const int* in_sizes, int n_in,
                              void* d_out, int out_size, void* d_ws, size_t ws_size,
                              hipStream_t stream) {
    const float* x      = (const float*)d_in[0];
    const int*   ei     = (const int*)d_in[1];   // [2,E]: src then dst
    const float* W_in   = (const float*)d_in[2];
    const float* b_in   = (const float*)d_in[3];
    const float* W_hid  = (const float*)d_in[4]; // [L,256,256]
    const float* b_hid  = (const float*)d_in[5]; // [L,256]
    const float* W_out  = (const float*)d_in[6]; // [256,32]
    const float* b_out  = (const float*)d_in[7];
    float* out = (float*)d_out;

    const int H = 256, Cc = 32;
    const int N = in_sizes[0] / H;    // 50000
    const int E = in_sizes[1] / 2;    // 800000

    // workspace layout (256B aligned)
    char* w = (char*)d_ws;
    size_t off = 0;
    auto take = [&](size_t bytes) -> void* {
        void* p = w + off;
        off += (bytes + 255) & ~(size_t)255;
        return p;
    };
    float*  deg_sn = (float*)take((size_t)N * 4);          // deg -> self_norm
    float*  dinv   = (float*)take((size_t)N * 4);
    float*  norme  = (float*)take((size_t)E * 4);
    bf16_t* Bp_in  = (bf16_t*)take((size_t)H * H * 2);     // packed weights
    bf16_t* Bp_h0  = (bf16_t*)take((size_t)H * H * 2);
    bf16_t* Bp_h1  = (bf16_t*)take((size_t)H * H * 2);
    bf16_t* Bp_o   = (bf16_t*)take((size_t)H * Cc * 2);
    bf16_t* Apack  = (bf16_t*)take((size_t)N * H * 2);     // packed activations
    float*  t      = (float*)take((size_t)N * H * 4);
    float*  agg    = (float*)take((size_t)N * H * 4);

    // --- graph normalization ---
    k_fill<<<cdiv(N, 256), 256, 0, stream>>>(deg_sn, 2.0f, N);            // A_hat = A + 2I
    k_count_deg<<<cdiv(E, 256), 256, 0, stream>>>(ei + E, deg_sn, E);
    k_norm<<<cdiv(N, 256), 256, 0, stream>>>(deg_sn, dinv, N);
    k_edge_norm<<<cdiv(E, 256), 256, 0, stream>>>(ei, dinv, norme, E);

    // --- pack weights into B-fragment order (once) ---
    const int wthreads = (H / 16) * 8 * 32;   // 4096 per 256x256 matrix
    const int othreads = (Cc / 16) * 8 * 32;  // 512 for 256x32
    k_pack_b<<<cdiv(wthreads, 256), 256, 0, stream>>>(W_in,          Bp_in, H,  wthreads);
    k_pack_b<<<cdiv(wthreads, 256), 256, 0, stream>>>(W_hid,         Bp_h0, H,  wthreads);
    k_pack_b<<<cdiv(wthreads, 256), 256, 0, stream>>>(W_hid + H * H, Bp_h1, H,  wthreads);
    k_pack_b<<<cdiv(othreads, 256), 256, 0, stream>>>(W_out,         Bp_o,  Cc, othreads);

    // --- pack input activations (no relu) ---
    const int athreads = (N / 16) * 8 * 32;   // 800000
    k_pack_a<<<cdiv(athreads, 256), 256, 0, stream>>>(x, Apack, athreads, 0);

    const int total256 = N * H;
    auto hidden_layer = [&](const bf16_t* Bl, const float* bl) {
        gemm_wmma<256, 4><<<dim3(cdiv(N, 16), 4), 128, 0, stream>>>(Apack, Bl, t);
        k_agg_init<256><<<cdiv(total256, 256), 256, 0, stream>>>(t, deg_sn, bl, agg, total256);
        k_scatter<256><<<cdiv(E, 4), 256, 0, stream>>>(t, ei, norme, agg, E);
        k_pack_a<<<cdiv(athreads, 256), 256, 0, stream>>>(agg, Apack, athreads, 1); // ReLU + repack
    };

    hidden_layer(Bp_in, b_in);            // input layer
    hidden_layer(Bp_h0, b_hid);           // hidden 0
    hidden_layer(Bp_h1, b_hid + H);       // hidden 1

    // output layer (NC=32): 2 waves per block, then tanh into d_out
    const int total32 = N * Cc;
    gemm_wmma<32, 2><<<dim3(cdiv(N, 16), 1), 64, 0, stream>>>(Apack, Bp_o, t);
    k_agg_init<32><<<cdiv(total32, 256), 256, 0, stream>>>(t, deg_sn, b_out, agg, total32);
    k_scatter<32><<<cdiv(E, 32), 256, 0, stream>>>(t, ei, norme, agg, E);
    k_tanh<<<cdiv(total32, 256), 256, 0, stream>>>(agg, out, total32);
}
